// MultiHeadSelfAttention_89172110999734
// MI455X (gfx1250) — matrix operand order repaired and checked by reading, not compiled, not run on hardware
//
#include <hip/hip_runtime.h>

typedef __attribute__((ext_vector_type(16))) __bf16 v16bf;
typedef __attribute__((ext_vector_type(8)))  __bf16 v8bf;
typedef __attribute__((ext_vector_type(8)))  float  v8f;

#define D_MODEL 1024
#define SEQ     2048
#define NBATCH  2
#define NHEADS  16
#define HEADDIM 64
#define MTOT    (NBATCH * SEQ)   // 4096

static __device__ __forceinline__ __bf16 f2bf(float f) {
  unsigned u = __builtin_bit_cast(unsigned, f);
  unsigned r = (u + 0x7FFFu + ((u >> 16) & 1u)) >> 16;
  unsigned short s = (unsigned short)r;
  return __builtin_bit_cast(__bf16, s);
}

static __device__ __forceinline__ v8f vzero8() {
  v8f z;
#pragma unroll
  for (int i = 0; i < 8; ++i) z[i] = 0.0f;
  return z;
}

// A-fragment layout: identity (linear K) so lane(half) reads 16 contiguous K
// values at half*16, matching the contiguous B fragments.
static __device__ __forceinline__ int swz32(int k) {
  return k;
}

static __device__ __forceinline__ v8f wmma_bf16(v16bf a, v16bf b, v8f c) {
  return __builtin_amdgcn_wmma_f32_16x16x32_bf16(false, a, false, b, (short)0, c,
                                                 false, false);
}

// ---------------------------------------------------------------- converts
__global__ __launch_bounds__(256)
void cvt_f32_to_bf16(const float* __restrict__ in, __bf16* __restrict__ out, int n) {
  int i = (blockIdx.x * 256 + threadIdx.x) * 4;
  if (i >= n) return;
  float4 v = *(const float4*)(in + i);
  out[i + 0] = f2bf(v.x);
  out[i + 1] = f2bf(v.y);
  out[i + 2] = f2bf(v.z);
  out[i + 3] = f2bf(v.w);
}

// f32 -> bf16 with per-row A-fragment swizzle (row length Kdim, multiple of 32)
__global__ __launch_bounds__(256)
void cvt_f32_to_bf16_swz(const float* __restrict__ in, __bf16* __restrict__ out,
                         int n, int Kdim) {
  int i = (blockIdx.x * 256 + threadIdx.x) * 4;
  if (i >= n) return;
  float4 v = *(const float4*)(in + i);
  const int m = i / Kdim;
  const int k = i - m * Kdim;
  __bf16* o = out + (size_t)m * Kdim + swz32(k);  // low 3 bits preserved
  o[0] = f2bf(v.x);
  o[1] = f2bf(v.y);
  o[2] = f2bf(v.z);
  o[3] = f2bf(v.w);
}

// ---------------------------------------------------------------- GEMM
// C[M,N] = A[M,K] @ W[N,K]^T + bias   (torch Linear), A stored swizzled.
// OUTMODE 0: bf16 row-major [M,N]               (K matrix: B-side consumer)
// OUTMODE 1: bf16 transposed-per-batch [b,n,s]  (V^T: B-side consumer)
// OUTMODE 2: f32 row-major [M,N]                (final output)
// OUTMODE 3: bf16 row-major, A-swizzled columns (Q: A-side consumer)
template <int OUTMODE>
__global__ __launch_bounds__(256)
void gemm_xwt(const __bf16* __restrict__ A, const __bf16* __restrict__ W,
              const float* __restrict__ bias, void* __restrict__ outv,
              int Mdim, int Ndim, int Kdim) {
  const int lane = threadIdx.x & 31;
  const int wid  = threadIdx.x >> 5;
  const int tilesN = Ndim / 64;
  const int wg = blockIdx.x * 8 + wid;
  const int mbase = (wg / tilesN) * 32;
  const int nbase = (wg % tilesN) * 64;
  const int half = lane >> 4;
  const int lm   = lane & 15;
  const int ka = half * 16;   // A-frag offset (swizzled layout)
  const int kb = half * 16;   // B-frag K offset

  v8f acc[2][4];
#pragma unroll
  for (int mi = 0; mi < 2; ++mi)
#pragma unroll
    for (int ni = 0; ni < 4; ++ni) acc[mi][ni] = vzero8();

  for (int kk = 0; kk < Kdim; kk += 32) {
    v16bf afrag[2], wfrag[4];
#pragma unroll
    for (int mi = 0; mi < 2; ++mi) {
      const __bf16* ap = A + (size_t)(mbase + mi * 16 + lm) * Kdim + kk + ka;
      afrag[mi] = *(const v16bf*)ap;   // 16 contiguous (swizzled) values
    }
#pragma unroll
    for (int ni = 0; ni < 4; ++ni) {
      const __bf16* wp = W + (size_t)(nbase + ni * 16 + lm) * Kdim + kk + kb;
      wfrag[ni] = *(const v16bf*)wp;   // 16 contiguous K values
    }
#pragma unroll
    for (int mi = 0; mi < 2; ++mi)
#pragma unroll
      for (int ni = 0; ni < 4; ++ni)
        acc[mi][ni] = wmma_bf16(afrag[mi], wfrag[ni], acc[mi][ni]);
  }

#pragma unroll
  for (int mi = 0; mi < 2; ++mi) {
#pragma unroll
    for (int ni = 0; ni < 4; ++ni) {
      const int n = nbase + ni * 16 + lm;
      const float bv = bias[n];
#pragma unroll
      for (int j = 0; j < 8; ++j) {
        const int m = mbase + mi * 16 + j + 8 * half;
        const float v = acc[mi][ni][j] + bv;
        if (OUTMODE == 0) {
          ((__bf16*)outv)[(size_t)m * Ndim + n] = f2bf(v);
        } else if (OUTMODE == 1) {
          const int bb = m / SEQ, ss = m % SEQ;
          ((__bf16*)outv)[((size_t)(bb * Ndim + n)) * SEQ + ss] = f2bf(v);
        } else if (OUTMODE == 2) {
          ((float*)outv)[(size_t)m * Ndim + n] = v;
        } else {
          ((__bf16*)outv)[(size_t)m * Ndim + swz32(n)] = f2bf(v);
        }
      }
    }
  }
}

// ---------------------------------------------------------------- attention
// One wave per 16-row q tile of one (batch, head). Flash-style online softmax,
// kv chunks of 32 (causal => stop at qbase+16). Q and O use swizzled A layout.
__global__ __launch_bounds__(256)
void attn_fwd(const __bf16* __restrict__ Q, const __bf16* __restrict__ K,
              const __bf16* __restrict__ Vt, __bf16* __restrict__ O) {
  __shared__ __bf16 ldsP[8][16 * 32];   // per-wave 16x32 P tile, swizzled cols
  const int lane = threadIdx.x & 31;
  const int wid  = threadIdx.x >> 5;
  const int wg = blockIdx.x * 8 + wid;
  const int qt = wg & 127;          // SEQ/16 = 128
  const int h  = (wg >> 7) & 15;
  const int b  = wg >> 11;
  const int qbase = qt * 16;
  const int half = lane >> 4;
  const int lm   = lane & 15;
  const int ka = half * 16, kb = half * 16;
  const int plo = lm;   // pos of col lm
  // pos of col 16+lm is plo + 16

  // Q A-fragments (swizzled layout): contiguous 16 values per lane
  const __bf16* qp = Q + (size_t)(b * SEQ + qbase + lm) * D_MODEL + h * HEADDIM;
  const v16bf aQ0 = *(const v16bf*)(qp + ka);
  const v16bf aQ1 = *(const v16bf*)(qp + 32 + ka);

  v8f outAcc[4];
#pragma unroll
  for (int t = 0; t < 4; ++t) outAcc[t] = vzero8();
  float mrow[8], lrow[8];
#pragma unroll
  for (int j = 0; j < 8; ++j) { mrow[j] = -1e30f; lrow[j] = 0.0f; }

  for (int kvb = 0; kvb < qbase + 16; kvb += 32) {
    // K B-fragments: lane holds kv row (kvb + sub*16 + lm), contiguous d
    const __bf16* kp0 = K + (size_t)(b * SEQ + kvb + lm) * D_MODEL + h * HEADDIM;
    const __bf16* kp1 = kp0 + (size_t)16 * D_MODEL;
    const v16bf bK0lo = *(const v16bf*)(kp0 + kb);
    const v16bf bK0hi = *(const v16bf*)(kp0 + 32 + kb);
    const v16bf bK1lo = *(const v16bf*)(kp1 + kb);
    const v16bf bK1hi = *(const v16bf*)(kp1 + 32 + kb);

    v8f s0 = wmma_bf16(aQ0, bK0lo, vzero8());
    s0     = wmma_bf16(aQ1, bK0hi, s0);
    v8f s1 = wmma_bf16(aQ0, bK1lo, vzero8());
    s1     = wmma_bf16(aQ1, bK1hi, s1);

#pragma unroll
    for (int j = 0; j < 8; ++j) {
      const int qrow = qbase + j + 8 * half;
      const int kv0 = kvb + lm, kv1 = kv0 + 16;
      float x0 = (kv0 <= qrow) ? s0[j] * 0.125f : -10000.0f;  // 1/sqrt(64)
      float x1 = (kv1 <= qrow) ? s1[j] * 0.125f : -10000.0f;
      float mx = fmaxf(x0, x1);
#pragma unroll
      for (int o = 1; o < 16; o <<= 1) mx = fmaxf(mx, __shfl_xor(mx, o));
      const float mn = fmaxf(mrow[j], mx);
      const float alpha = __expf(mrow[j] - mn);
      mrow[j] = mn;
      const float p0 = __expf(x0 - mn);
      const float p1 = __expf(x1 - mn);
      float rs = p0 + p1;
#pragma unroll
      for (int o = 1; o < 16; o <<= 1) rs += __shfl_xor(rs, o);
      lrow[j] = lrow[j] * alpha + rs;
#pragma unroll
      for (int t = 0; t < 4; ++t) outAcc[t][j] *= alpha;
      // stash P (C layout) into LDS at swizzled column positions
      const int prow = (j + 8 * half) * 32;
      ldsP[wid][prow + plo]     = f2bf(p0);
      ldsP[wid][prow + plo + 16] = f2bf(p1);
    }

    // P as A fragment: one contiguous 32B LDS read (same-wave DS is in-order)
    const v16bf aP = *(const v16bf*)&ldsP[wid][lm * 32 + ka];

    // V B-fragments from Vt[b*1024 + d, s]: contiguous in s
#pragma unroll
    for (int t = 0; t < 4; ++t) {
      const __bf16* vp = Vt + (size_t)(b * D_MODEL + h * HEADDIM + t * 16 + lm) * SEQ
                         + kvb + kb;
      const v16bf bV = *(const v16bf*)vp;
      outAcc[t] = wmma_bf16(aP, bV, outAcc[t]);
    }
  }

  // epilogue: store O in swizzled A layout (consumed by final projection GEMM)
#pragma unroll
  for (int j = 0; j < 8; ++j) {
    const float inv = 1.0f / lrow[j];
    const int m = b * SEQ + qbase + j + 8 * half;
#pragma unroll
    for (int t = 0; t < 4; ++t) {
      const int n = h * HEADDIM + t * 16 + lm;
      O[(size_t)m * D_MODEL + swz32(n)] = f2bf(outAcc[t][j] * inv);
    }
  }
}

// ---------------------------------------------------------------- launcher
extern "C" void kernel_launch(void* const* d_in, const int* in_sizes, int n_in,
                              void* d_out, int out_size, void* d_ws, size_t ws_size,
                              hipStream_t stream) {
  const float* x  = (const float*)d_in[0];
  const float* Wq = (const float*)d_in[1];
  const float* bq = (const float*)d_in[2];
  const float* Wk = (const float*)d_in[3];
  const float* bk = (const float*)d_in[4];
  const float* Wv = (const float*)d_in[5];
  const float* bv = (const float*)d_in[6];
  const float* Wo = (const float*)d_in[7];
  const float* bo = (const float*)d_in[8];

  char* ws = (char*)d_ws;
  const size_t SZ_X = (size_t)MTOT * D_MODEL * 2;          // 8 MB bf16
  const size_t SZ_W = (size_t)D_MODEL * D_MODEL * 2;       // 2 MB bf16
  __bf16* xb  = (__bf16*)(ws);                             // swizzled A
  __bf16* Wqb = (__bf16*)(ws + SZ_X);
  __bf16* Wkb = (__bf16*)(ws + SZ_X + SZ_W);
  __bf16* Wvb = (__bf16*)(ws + SZ_X + 2 * SZ_W);
  __bf16* Wob = (__bf16*)(ws + SZ_X + 3 * SZ_W);
  __bf16* Qb  = (__bf16*)(ws + SZ_X + 4 * SZ_W);           // swizzled A
  __bf16* Kb  = (__bf16*)(ws + 2 * SZ_X + 4 * SZ_W);       // linear (B-side)
  __bf16* Vtb = (__bf16*)(ws + 3 * SZ_X + 4 * SZ_W);       // transposed (B-side)
  __bf16* Ob  = (__bf16*)(ws + 4 * SZ_X + 4 * SZ_W);       // swizzled A

  const int nX = MTOT * D_MODEL;       // 4194304
  const int nW = D_MODEL * D_MODEL;    // 1048576

  cvt_f32_to_bf16_swz<<<nX / 4 / 256, 256, 0, stream>>>(x, xb, nX, D_MODEL);
  cvt_f32_to_bf16<<<nW / 4 / 256, 256, 0, stream>>>(Wq, Wqb, nW);
  cvt_f32_to_bf16<<<nW / 4 / 256, 256, 0, stream>>>(Wk, Wkb, nW);
  cvt_f32_to_bf16<<<nW / 4 / 256, 256, 0, stream>>>(Wv, Wvb, nW);
  cvt_f32_to_bf16<<<nW / 4 / 256, 256, 0, stream>>>(Wo, Wob, nW);

  const int gemmBlocks = (MTOT / 32) * (D_MODEL / 64) / 8;  // 256
  gemm_xwt<3><<<gemmBlocks, 256, 0, stream>>>(xb, Wqb, bq, (void*)Qb,
                                              MTOT, D_MODEL, D_MODEL);
  gemm_xwt<0><<<gemmBlocks, 256, 0, stream>>>(xb, Wkb, bk, (void*)Kb,
                                              MTOT, D_MODEL, D_MODEL);
  gemm_xwt<1><<<gemmBlocks, 256, 0, stream>>>(xb, Wvb, bv, (void*)Vtb,
                                              MTOT, D_MODEL, D_MODEL);

  const int attnBlocks = (NBATCH * NHEADS * (SEQ / 16)) / 8;  // 512
  attn_fwd<<<attnBlocks, 256, 0, stream>>>(Qb, Kb, Vtb, Ob);

  gemm_xwt<2><<<gemmBlocks, 256, 0, stream>>>(Ob, Wob, bo, d_out,
                                              MTOT, D_MODEL, D_MODEL);
}
